// EdgeNetDynamic_7456063226154
// MI455X (gfx1250) — matrix operand order repaired and checked by reading, not compiled, not run on hardware
//
#include <hip/hip_runtime.h>
#include <hip/hip_bf16.h>

typedef __attribute__((ext_vector_type(16))) _Float16 v16h;
typedef __attribute__((ext_vector_type(8)))  _Float16 v8h;
typedef __attribute__((ext_vector_type(8)))  float    v8f;

#define WMMA_F16(A, B, C) \
  __builtin_amdgcn_wmma_f32_16x16x32_f16(false, (A), false, (B), (short)0, (C), false, false)

// ---------------------------------------------------------------------------
// BatchNorm statistics: sum / sumsq per feature (F=4) into stats[0..7].
// ---------------------------------------------------------------------------
__global__ __launch_bounds__(256)
void bn_stats_kernel(const float* __restrict__ X, float* __restrict__ stats, int nNodes) {
  float s[4] = {0.f, 0.f, 0.f, 0.f};
  float q[4] = {0.f, 0.f, 0.f, 0.f};
  const int stride = gridDim.x * blockDim.x;
  for (int i = blockIdx.x * blockDim.x + threadIdx.x; i < nNodes; i += stride) {
    float4 v = ((const float4*)X)[i];
    s[0] += v.x; s[1] += v.y; s[2] += v.z; s[3] += v.w;
    q[0] = fmaf(v.x, v.x, q[0]); q[1] = fmaf(v.y, v.y, q[1]);
    q[2] = fmaf(v.z, v.z, q[2]); q[3] = fmaf(v.w, v.w, q[3]);
  }
#pragma unroll
  for (int off = 16; off > 0; off >>= 1) {
#pragma unroll
    for (int f = 0; f < 4; ++f) {
      s[f] += __shfl_down(s[f], off, 32);
      q[f] += __shfl_down(q[f], off, 32);
    }
  }
  if ((threadIdx.x & 31) == 0) {
#pragma unroll
    for (int f = 0; f < 4; ++f) {
      atomicAdd(&stats[f], s[f]);
      atomicAdd(&stats[4 + f], q[f]);
    }
  }
}

// ---------------------------------------------------------------------------
// B-operand tile loader: B is [K=32 x N=16] f16. Lane L holds column
// n = (L&15); half-wave h=(L>>4): element i -> k = 8h + i for i < 8,
// k = 16 + 8h + (i - 8) for i >= 8.  Rows k >= kdim (or invalid n) are zero.
// ---------------------------------------------------------------------------
__device__ __forceinline__
v16h load_btile(const float* __restrict__ W, int ld, int kdim, int n, int hf, bool nvalid) {
  v16h b;
#pragma unroll
  for (int i = 0; i < 16; ++i) {
    int k = (i < 8) ? (8 * hf + i) : (16 + 8 * hf + (i - 8));
    float v = (nvalid && k < kdim) ? W[k * ld + n] : 0.f;
    b[i] = (_Float16)v;
  }
  return b;
}

// ---------------------------------------------------------------------------
// One DynamicEdgeConv layer for one graph per workgroup (256 thr = 8 waves).
// FIN: input feature dim, FOUT: output dim, RELU: final relu, NORM: apply BN.
// ---------------------------------------------------------------------------
template <int FIN, int FOUT, bool RELU, bool NORM>
__global__ __launch_bounds__(256)
void edgeconv_kernel(const float* __restrict__ X,
                     const float* __restrict__ stats,
                     const float* __restrict__ gamma,
                     const float* __restrict__ beta,
                     const float* __restrict__ W1, const float* __restrict__ B1,
                     const float* __restrict__ W2, const float* __restrict__ B2,
                     const float* __restrict__ W3, const float* __restrict__ B3,
                     float* __restrict__ OUT, float invCount) {
  constexpr int N = 256, KNN = 3, BIG = 32;
  constexpr int NE = N * KNN;        // 768 edges per graph
  constexpr int NTILE = NE / 16;     // 48 row-tiles of 16 edges

  __shared__ float hp[N * FIN];                       // points (normalized)
  __shared__ int   nbr[N * KNN];                      // knn indices
  __shared__ __align__(16) _Float16 stage[8][16 * 32]; // per-wave relayout tile
  __shared__ float msg[NE * FOUT];                    // per-edge MLP output

  const int tid = threadIdx.x;
  const size_t base = (size_t)blockIdx.x * N;

  // ---- 1) load (+ BatchNorm) this graph's points into LDS -----------------
  {
    const float* xp = X + (base + tid) * FIN;
#pragma unroll
    for (int f = 0; f < FIN; ++f) {
      float v = xp[f];
      if constexpr (NORM) {
        float mu  = stats[f] * invCount;
        float var = stats[4 + f] * invCount - mu * mu;
        v = (v - mu) * rsqrtf(var + 1e-5f) * gamma[f] + beta[f];
      }
      hp[tid * FIN + f] = v;
    }
  }
  __syncthreads();

  // ---- 2) kNN (k=3, includes self; ties -> lowest index) ------------------
  {
    float xi[FIN];
#pragma unroll
    for (int f = 0; f < FIN; ++f) xi[f] = hp[tid * FIN + f];
    float d0 = 3.4e38f, d1 = 3.4e38f, d2 = 3.4e38f;
    int   i0 = 0, i1 = 0, i2 = 0;
    for (int m = 0; m < N; ++m) {
      float d = 0.f;
#pragma unroll
      for (int f = 0; f < FIN; ++f) {
        float t = xi[f] - hp[m * FIN + f];
        d = fmaf(t, t, d);
      }
      if (d < d0)      { d2 = d1; i2 = i1; d1 = d0; i1 = i0; d0 = d; i0 = m; }
      else if (d < d1) { d2 = d1; i2 = i1; d1 = d;  i1 = m; }
      else if (d < d2) { d2 = d;  i2 = m; }
    }
    nbr[tid * 3 + 0] = i0; nbr[tid * 3 + 1] = i1; nbr[tid * 3 + 2] = i2;
  }
  __syncthreads();

  // ---- 3) edge MLP via WMMA ----------------------------------------------
  const int lane = tid & 31;
  const int wv   = tid >> 5;
  const int hf   = lane >> 4;   // half-wave
  const int mrow = lane & 15;

  // Weights as B operands (K padded to 32, N padded to 16 where needed).
  v16h b1_0 = load_btile(W1, BIG,  2 * FIN, mrow,      hf, true);
  v16h b1_1 = load_btile(W1, BIG,  2 * FIN, mrow + 16, hf, true);
  v16h b2_0 = load_btile(W2, BIG,  BIG,     mrow,      hf, true);
  v16h b2_1 = load_btile(W2, BIG,  BIG,     mrow + 16, hf, true);
  v16h b3_0 = load_btile(W3, FOUT, BIG,     mrow,      hf, mrow < FOUT);

  const float bias1_0 = B1[mrow], bias1_1 = B1[mrow + 16];
  const float bias2_0 = B2[mrow], bias2_1 = B2[mrow + 16];
  const float bias3   = (mrow < FOUT) ? B3[mrow] : 0.f;

  _Float16* st = &stage[wv][0];

  for (int t = wv; t < NTILE; t += 8) {
    // A1 tile: 16 edges x 32 (cols >= 2*FIN are zero). A-layout per ISA:
    // lanes 0-15 hold K 0-7 / 16-23, lanes 16-31 hold K 8-15 / 24-31.
    const int e  = t * 16 + mrow;
    const int p  = e / 3;
    const int kk = e - p * 3;
    const int j  = nbr[p * 3 + kk];
    v16h a;
#pragma unroll
    for (int i = 0; i < 16; ++i) a[i] = (_Float16)0.f;
    if (hf == 0) {
#pragma unroll
      for (int f = 0; f < FIN; ++f) {
        float xi = hp[p * FIN + f];
        float xj = hp[j * FIN + f];
        a[f]       = (_Float16)xi;          // K = f
        a[FIN + f] = (_Float16)(xj - xi);   // K = FIN + f
      }
    }

    // layer 1: [16x32] x [32x32] -> [16x32]
    v8f c0 = {}, c1 = {};
    c0 = WMMA_F16(a, b1_0, c0);
    c1 = WMMA_F16(a, b1_1, c1);
#pragma unroll
    for (int r = 0; r < 8; ++r) {
      int m = r + 8 * hf;
      st[m * 32 + mrow]      = (_Float16)fmaxf(c0[r] + bias1_0, 0.f);
      st[m * 32 + 16 + mrow] = (_Float16)fmaxf(c1[r] + bias1_1, 0.f);
    }
    __builtin_amdgcn_wave_barrier();
    v8h lo = *(v8h*)&st[mrow * 32 + 8 * hf];
    v8h hi = *(v8h*)&st[mrow * 32 + 16 + 8 * hf];
    v16h a2;
#pragma unroll
    for (int i = 0; i < 8; ++i) { a2[i] = lo[i]; a2[8 + i] = hi[i]; }
    __builtin_amdgcn_wave_barrier();

    // layer 2: [16x32] x [32x32] -> [16x32]
    v8f h0 = {}, h1 = {};
    h0 = WMMA_F16(a2, b2_0, h0);
    h1 = WMMA_F16(a2, b2_1, h1);
#pragma unroll
    for (int r = 0; r < 8; ++r) {
      int m = r + 8 * hf;
      st[m * 32 + mrow]      = (_Float16)fmaxf(h0[r] + bias2_0, 0.f);
      st[m * 32 + 16 + mrow] = (_Float16)fmaxf(h1[r] + bias2_1, 0.f);
    }
    __builtin_amdgcn_wave_barrier();
    lo = *(v8h*)&st[mrow * 32 + 8 * hf];
    hi = *(v8h*)&st[mrow * 32 + 16 + 8 * hf];
    v16h a3;
#pragma unroll
    for (int i = 0; i < 8; ++i) { a3[i] = lo[i]; a3[8 + i] = hi[i]; }
    __builtin_amdgcn_wave_barrier();

    // layer 3: [16x32] x [32x16] -> [16x16], only cols < FOUT valid
    v8f o = {};
    o = WMMA_F16(a3, b3_0, o);
    if (mrow < FOUT) {
#pragma unroll
      for (int r = 0; r < 8; ++r) {
        int m = r + 8 * hf;
        float v = o[r] + bias3;
        if (RELU) v = fmaxf(v, 0.f);
        msg[(t * 16 + m) * FOUT + mrow] = v;
      }
    }
  }
  __syncthreads();

  // ---- 4) mean over K=3 and store ----------------------------------------
  {
    const float inv3 = 1.f / 3.f;
    float* op = OUT + (base + tid) * FOUT;
#pragma unroll
    for (int f = 0; f < FOUT; ++f) {
      float s = msg[(tid * 3 + 0) * FOUT + f] +
                msg[(tid * 3 + 1) * FOUT + f] +
                msg[(tid * 3 + 2) * FOUT + f];
      op[f] = s * inv3;
    }
  }
}

// ---------------------------------------------------------------------------
extern "C" void kernel_launch(void* const* d_in, const int* in_sizes, int n_in,
                              void* d_out, int out_size, void* d_ws, size_t ws_size,
                              hipStream_t stream) {
  const float* x     = (const float*)d_in[0];
  const float* gamma = (const float*)d_in[1];
  const float* beta  = (const float*)d_in[2];
  const float* ew1 = (const float*)d_in[3];
  const float* eb1 = (const float*)d_in[4];
  const float* ew2 = (const float*)d_in[5];
  const float* eb2 = (const float*)d_in[6];
  const float* ew3 = (const float*)d_in[7];
  const float* eb3 = (const float*)d_in[8];
  const float* dw1 = (const float*)d_in[9];
  const float* db1 = (const float*)d_in[10];
  const float* dw2 = (const float*)d_in[11];
  const float* db2 = (const float*)d_in[12];
  const float* dw3 = (const float*)d_in[13];
  const float* db3 = (const float*)d_in[14];

  const int B = 1024, N = 256;
  float* stats  = (float*)d_ws;          // 8 floats: sum[4], sumsq[4]
  float* hidden = (float*)d_ws + 16;     // [B, N, 2] = 2 MB
  const float invCount = 1.0f / (float)(B * N);

  hipMemsetAsync(stats, 0, 8 * sizeof(float), stream);
  bn_stats_kernel<<<256, 256, 0, stream>>>(x, stats, B * N);

  // encoder: F=4 -> edge 8 -> 32 -> 32 -> 2, final relu, with batchnorm
  edgeconv_kernel<4, 2, true, true><<<B, 256, 0, stream>>>(
      x, stats, gamma, beta, ew1, eb1, ew2, eb2, ew3, eb3, hidden, invCount);

  // decoder: F=2 -> edge 4 -> 32 -> 32 -> 4, no final relu, no batchnorm
  edgeconv_kernel<2, 4, false, false><<<B, 256, 0, stream>>>(
      hidden, nullptr, nullptr, nullptr, dw1, db1, dw2, db2, dw3, db3,
      (float*)d_out, 0.f);
}
